// GTN_1554778161476
// MI455X (gfx1250) — matrix operand order repaired and checked by reading, not compiled, not run on hardware
//
#include <hip/hip_runtime.h>
#include <hip/hip_bf16.h>
#include <stdint.h>

// ---------------------------------------------------------------------------
// Problem constants
// ---------------------------------------------------------------------------
static constexpr int    NDIM  = 2048;
static constexpr int    EDIM  = 5;
static constexpr int    CDIM  = 2;
static constexpr int    WIN   = 256;
static constexpr int    WOUT  = 64;
static constexpr size_t NN    = (size_t)NDIM * NDIM;   // 4194304 = 2^22

// ---------------------------------------------------------------------------
// Workspace layout (bytes). Aliased regions noted.
// ---------------------------------------------------------------------------
static constexpr size_t SW_OFF   = 0;                       // 3*C*E floats (softmax weights)
static constexpr size_t ABF_OFF  = 512;                     // bf16 a   [C][N][N]   (16 MB)
static constexpr size_t BT_OFF   = ABF_OFF + 2 * NN * 2;    // bf16 b^T [C][N][N]
static constexpr size_t GT_OFF   = BT_OFF  + 2 * NN * 2;    // bf16 g^T [C][N][N]
static constexpr size_t H_OFF    = GT_OFF  + 2 * NN * 2;    // f32 H / H2 [C][N][N] (33.5 MB, reused)
static constexpr size_t HNBF_OFF = H_OFF   + 2 * NN * 4;    // bf16 Hn [C][N][N]
static constexpr size_t CINV_OFF = HNBF_OFF + 2 * NN * 2;   // f32 [C][N] col-inverse
static constexpr size_t PART_OFF = CINV_OFF + 2 * NDIM * 4; // f32 [16][C*N] partial col sums
// Aliases (ABF region is dead after the first GEMM):
static constexpr size_t XW_OFF   = ABF_OFF;                          // f32 [N][64]
static constexpr size_t XC_OFF   = ABF_OFF + (size_t)NDIM * WOUT * 4;// f32 [C][N][64]

// ---------------------------------------------------------------------------
// WMMA vector types
// ---------------------------------------------------------------------------
typedef __bf16        v16bf __attribute__((ext_vector_type(16)));
typedef float         v8f   __attribute__((ext_vector_type(8)));
typedef unsigned int  u32x4 __attribute__((ext_vector_type(4)));

union FragU { v16bf bf; u32x4 q[2]; };

__device__ __forceinline__ unsigned short f2bf(float f) {
    unsigned int u = __float_as_uint(f);
    u += 0x7FFFu + ((u >> 16) & 1u);          // round-to-nearest-even
    return (unsigned short)(u >> 16);
}

// LDS byte offset of a generic (flat) shared pointer: per CDNA5 ISA §10.2 the
// flat->LDS mapping uses addr[31:0] as the wave-relative LDS address.
__device__ __forceinline__ unsigned lds_off(const void* p) {
    return (unsigned)(uintptr_t)p;
}

// ---------------------------------------------------------------------------
// Stage 0: softmax of the three [C,E] weight sets -> ws.sw[kind][c][e]
// ---------------------------------------------------------------------------
__global__ void k_softmax(const float* __restrict__ w0a,
                          const float* __restrict__ w0b,
                          const float* __restrict__ w1,
                          float* __restrict__ sw) {
    if (threadIdx.x != 0) return;
    const float* src[3] = { w0a, w0b, w1 };
    for (int kind = 0; kind < 3; ++kind) {
        for (int c = 0; c < CDIM; ++c) {
            float m = src[kind][c * EDIM];
            for (int e = 1; e < EDIM; ++e) m = fmaxf(m, src[kind][c * EDIM + e]);
            float s = 0.f, ex[EDIM];
            for (int e = 0; e < EDIM; ++e) { ex[e] = __expf(src[kind][c * EDIM + e] - m); s += ex[e]; }
            for (int e = 0; e < EDIM; ++e) sw[kind * 10 + c * EDIM + e] = ex[e] / s;
        }
    }
}

// ---------------------------------------------------------------------------
// Stage 1: weighted adjacencies. a row-major bf16; b,g stored TRANSPOSED
//          (N x K row-major) so WMMA B-fragments are contiguous loads.
// ---------------------------------------------------------------------------
__global__ __launch_bounds__(256) void k_stage1(const float* __restrict__ A,
                                                const float* __restrict__ sw,
                                                unsigned short* __restrict__ aBF,
                                                unsigned short* __restrict__ bT,
                                                unsigned short* __restrict__ gT) {
    int idx = blockIdx.x * 256 + threadIdx.x;       // 0 .. N*N-1
    int i = idx >> 11;
    int j = idx & (NDIM - 1);
    size_t base = (size_t)idx * EDIM;
    float e0 = A[base + 0], e1 = A[base + 1], e2 = A[base + 2],
          e3 = A[base + 3], e4 = A[base + 4];
#pragma unroll
    for (int c = 0; c < CDIM; ++c) {
        const float* s0 = sw + 0  + c * EDIM;
        const float* s1 = sw + 10 + c * EDIM;
        const float* s2 = sw + 20 + c * EDIM;
        float av = s0[0]*e0 + s0[1]*e1 + s0[2]*e2 + s0[3]*e3 + s0[4]*e4;
        float bv = s1[0]*e0 + s1[1]*e1 + s1[2]*e2 + s1[3]*e3 + s1[4]*e4;
        float gv = s2[0]*e0 + s2[1]*e1 + s2[2]*e2 + s2[3]*e3 + s2[4]*e4;
        size_t cNN  = (size_t)c * NN;
        size_t tIdx = cNN + (size_t)j * NDIM + i;   // transposed slot (stays in L2)
        aBF[cNN + idx] = f2bf(av);
        bT[tIdx]       = f2bf(bv);
        gT[tIdx]       = f2bf(gv);
    }
}

// ---------------------------------------------------------------------------
// Stage 2/4: batched bf16 GEMM, f32 accumulate via V_WMMA_F32_16X16X32_BF16.
//   C[z] (MxN, f32, row-major) = Amat[z] (MxK bf16 row-major)
//                              x Bt[z]^T (Bt is NxK bf16 row-major)
// Block tile 128x128, 8 waves (4 in M x 2 in N), each wave 32x64 = 8 frags.
// Operand staging uses GLOBAL_LOAD_ASYNC_TO_LDS_B128 (ASYNCcnt path) with
// explicit LDS double buffering so the async engine runs behind the WMMAs.
// ---------------------------------------------------------------------------
static constexpr int LDS_STRIDE = 40;   // 32 K-elements + 8 pad (bank spread)

__global__ __launch_bounds__(256) void k_gemm_bf16(const unsigned short* __restrict__ Aall,
                                                   const unsigned short* __restrict__ Btall,
                                                   float* __restrict__ Call) {
    __shared__ __align__(16) unsigned short ldsA[2][128 * LDS_STRIDE];
    __shared__ __align__(16) unsigned short ldsB[2][128 * LDS_STRIDE];

    const size_t cOff = (size_t)blockIdx.z * NN;
    const unsigned short* Ab = Aall  + cOff;
    const unsigned short* Bb = Btall + cOff;
    float*                Cb = Call  + cOff;

    const int blockM = blockIdx.y * 128;
    const int blockN = blockIdx.x * 128;
    const int tid    = threadIdx.x;
    const int wave   = tid >> 5;
    const int lane   = tid & 31;
    const int waveM  = wave & 3;    // 4 waves along M (32 rows each)
    const int waveN  = wave >> 2;   // 2 waves along N (64 cols each)
    const int lhalf  = lane >> 4;   // 0 | 1  (lane half)
    const int l15    = lane & 15;

    // This thread's two 16B staging chunks (fixed row/sub across iterations).
    const int ch0  = tid;                 // chunks 0..255
    const int ch1  = 256 + tid;           // chunks 256..511
    const int row0 = ch0 >> 2, sub0 = ch0 & 3;
    const int row1 = ch1 >> 2, sub1 = ch1 & 3;

    // Issue async (VGPR-bypassing) copies of the 128x32 bf16 A/B tiles.
    auto stage_async = [&](int k0, int buf) {
        unsigned offA0 = ((unsigned)(blockM + row0) * (unsigned)NDIM + (unsigned)(k0 + sub0 * 8)) * 2u;
        unsigned offA1 = ((unsigned)(blockM + row1) * (unsigned)NDIM + (unsigned)(k0 + sub1 * 8)) * 2u;
        unsigned offB0 = ((unsigned)(blockN + row0) * (unsigned)NDIM + (unsigned)(k0 + sub0 * 8)) * 2u;
        unsigned offB1 = ((unsigned)(blockN + row1) * (unsigned)NDIM + (unsigned)(k0 + sub1 * 8)) * 2u;
        unsigned la0 = lds_off(&ldsA[buf][row0 * LDS_STRIDE + sub0 * 8]);
        unsigned la1 = lds_off(&ldsA[buf][row1 * LDS_STRIDE + sub1 * 8]);
        unsigned lb0 = lds_off(&ldsB[buf][row0 * LDS_STRIDE + sub0 * 8]);
        unsigned lb1 = lds_off(&ldsB[buf][row1 * LDS_STRIDE + sub1 * 8]);
        asm volatile("global_load_async_to_lds_b128 %0, %1, %2"
                     :: "v"(la0), "v"(offA0), "s"(Ab) : "memory");
        asm volatile("global_load_async_to_lds_b128 %0, %1, %2"
                     :: "v"(la1), "v"(offA1), "s"(Ab) : "memory");
        asm volatile("global_load_async_to_lds_b128 %0, %1, %2"
                     :: "v"(lb0), "v"(offB0), "s"(Bb) : "memory");
        asm volatile("global_load_async_to_lds_b128 %0, %1, %2"
                     :: "v"(lb1), "v"(offB1), "s"(Bb) : "memory");
    };

    v8f zero = {0.f,0.f,0.f,0.f,0.f,0.f,0.f,0.f};
    v8f acc[2][4];
#pragma unroll
    for (int m = 0; m < 2; ++m)
#pragma unroll
        for (int n = 0; n < 4; ++n) acc[m][n] = zero;

    // Prologue: stage tile 0, wait for the async engine, make it visible.
    stage_async(0, 0);
    asm volatile("s_wait_asynccnt 0x0" ::: "memory");
    __syncthreads();

    for (int k0 = 0; k0 < NDIM; k0 += 32) {
        const int p = (k0 >> 5) & 1;
        if (k0 + 32 < NDIM)
            stage_async(k0 + 32, p ^ 1);   // prefetch next tile behind the math

        // A fragment: lane l -> row M=l&15; K = {base..base+7, 16+base..16+base+7}, base=(l>>4)*8
        FragU aF[2];
#pragma unroll
        for (int m = 0; m < 2; ++m) {
            int r  = waveM * 32 + m * 16 + l15;
            int kb = lhalf * 8;
            aF[m].q[0] = *(const u32x4*)(&ldsA[p][r * LDS_STRIDE + kb]);
            aF[m].q[1] = *(const u32x4*)(&ldsA[p][r * LDS_STRIDE + 16 + kb]);
        }
        // B fragment (from transposed operand): lane l -> col N=l&15; K = {base..base+7, 16+base..16+base+7}, base=(l>>4)*8
        FragU bF[4];
#pragma unroll
        for (int n = 0; n < 4; ++n) {
            int r  = waveN * 64 + n * 16 + l15;
            int kb = lhalf * 8;
            bF[n].q[0] = *(const u32x4*)(&ldsB[p][r * LDS_STRIDE + kb]);
            bF[n].q[1] = *(const u32x4*)(&ldsB[p][r * LDS_STRIDE + 16 + kb]);
        }
#pragma unroll
        for (int m = 0; m < 2; ++m)
#pragma unroll
            for (int n = 0; n < 4; ++n)
                acc[m][n] = __builtin_amdgcn_wmma_f32_16x16x32_bf16(
                    false, aF[m].bf, false, bF[n].bf,
                    (short)0, acc[m][n], false, false);

        // Next buffer fully written (our asyncs done + everyone's barrier),
        // and every wave has consumed the current buffer (WMMA data deps).
        asm volatile("s_wait_asynccnt 0x0" ::: "memory");
        __syncthreads();
    }

    // D layout: VGPR v -> M = v + 8*(lane>>4), N = lane&15
#pragma unroll
    for (int m = 0; m < 2; ++m) {
#pragma unroll
        for (int n = 0; n < 4; ++n) {
            int col = blockN + waveN * 64 + n * 16 + l15;
#pragma unroll
            for (int v = 0; v < 8; ++v) {
                int rowg = blockM + waveM * 32 + m * 16 + v + 8 * lhalf;
                Cb[(size_t)rowg * NDIM + col] = acc[m][n][v];
            }
        }
    }
}

// ---------------------------------------------------------------------------
// Stage 3a: deterministic partial column sums (diag excluded).
//   grid.x = C*N/256 column groups, grid.y = 16 row chunks of 128.
// ---------------------------------------------------------------------------
__global__ __launch_bounds__(256) void k_colpart(const float* __restrict__ H,
                                                 float* __restrict__ part) {
    int idx = blockIdx.x * 256 + threadIdx.x;   // c*N + j
    int c = idx >> 11;
    int j = idx & (NDIM - 1);
    int i0 = blockIdx.y * 128;
    const float* Hc = H + (size_t)c * NN;
    float s = 0.f;
    for (int i = i0; i < i0 + 128; ++i) {
        float v = Hc[(size_t)i * NDIM + j];
        s += (i == j) ? 0.f : v;
    }
    part[(size_t)blockIdx.y * (CDIM * NDIM) + idx] = s;
}

// Stage 3b: reduce 16 partials -> inverse degree (ref: deg==0 -> 0)
__global__ __launch_bounds__(256) void k_colinv(const float* __restrict__ part,
                                                float* __restrict__ cinv) {
    int idx = blockIdx.x * 256 + threadIdx.x;   // c*N + j
    float s = 0.f;
    for (int ch = 0; ch < 16; ++ch) s += part[(size_t)ch * (CDIM * NDIM) + idx];
    cinv[idx] = (s == 0.f) ? 0.f : 1.f / s;
}

// Stage 3c: Hn[i][j] = (i==j ? 0 : H[i][j] * cinv[j]) -> bf16
__global__ __launch_bounds__(256) void k_scale_bf16(const float* __restrict__ H,
                                                    const float* __restrict__ cinv,
                                                    unsigned short* __restrict__ HnBF) {
    size_t idx = (size_t)blockIdx.x * 256 + threadIdx.x;   // 0 .. C*NN-1
    int c = (int)(idx >> 22);
    size_t rest = idx & (NN - 1);
    int i = (int)(rest >> 11);
    int j = (int)(rest & (NDIM - 1));
    float v = H[idx];
    v = (i == j) ? 0.f : v * cinv[c * NDIM + j];
    HnBF[idx] = f2bf(v);
}

// ---------------------------------------------------------------------------
// Stage 5: XW = X @ weight  (f32, tiny)
// ---------------------------------------------------------------------------
__global__ __launch_bounds__(64) void k_xw(const float* __restrict__ X,
                                           const float* __restrict__ W,
                                           float* __restrict__ XW) {
    int n = blockIdx.x;
    int k = threadIdx.x;
    const float* xr = X + (size_t)n * WIN;      // uniform -> scalar loads
    float s = 0.f;
    for (int i = 0; i < WIN; ++i) s += xr[i] * W[(size_t)i * WOUT + k];
    XW[(size_t)n * WOUT + k] = s;
}

// ---------------------------------------------------------------------------
// Stage 6: per channel GCN step.
//   Xc[c][i][k] = relu( deg_inv[i] * ( XW[i][k] + sum_{j!=i} H2[j][i]*XW[j][k] ) )
//   deg[i] = 1 + sum_{j!=i} H2[j][i]
// Lane owns column i (coalesced H2 row reads); wave owns a 16-wide k chunk.
// ---------------------------------------------------------------------------
__global__ __launch_bounds__(256) void k_gcn(const float* __restrict__ H2,
                                             const float* __restrict__ XW,
                                             float* __restrict__ Xc) {
    int w     = threadIdx.x >> 5;
    int lane  = threadIdx.x & 31;
    int kc    = w & 3;          // k chunk: 16 outputs
    int ihalf = w >> 2;         // which 32-column half of the block's 64 columns
    int c     = blockIdx.x >> 5;                         // 32 blocks per channel
    int i     = (blockIdx.x & 31) * 64 + ihalf * 32 + lane;
    const float* Hc = H2 + (size_t)c * NN;

    float acc[16];
#pragma unroll
    for (int k = 0; k < 16; ++k) acc[k] = 0.f;
    float deg = 0.f;

    for (int j = 0; j < NDIM; ++j) {
        float v = Hc[(size_t)j * NDIM + i];              // coalesced across lanes
        v = (j == i) ? 1.0f : v;                         // diag replaced by identity
        deg += v;
        const float* xr = XW + (size_t)j * WOUT + kc * 16;  // uniform -> scalar loads
#pragma unroll
        for (int k = 0; k < 16; ++k) acc[k] += v * xr[k];
    }
    float inv = (deg == 0.f) ? 0.f : 1.f / deg;
    float* op = Xc + ((size_t)(c * NDIM + i)) * WOUT + kc * 16;
#pragma unroll
    for (int k = 0; k < 16; ++k) {
        float r = acc[k] * inv;
        op[k] = r > 0.f ? r : 0.f;
    }
}

// ---------------------------------------------------------------------------
// Stage 7: out = concat_c(Xc) @ lin1_w + lin1_b   ([N,128] @ [128,64])
// ---------------------------------------------------------------------------
__global__ __launch_bounds__(64) void k_final(const float* __restrict__ Xc,
                                              const float* __restrict__ lw,
                                              const float* __restrict__ lb,
                                              float* __restrict__ out) {
    __shared__ float wsm[CDIM * WOUT * WOUT];   // 128*64 f32 = 32 KB
    int n = blockIdx.x;
    int o = threadIdx.x;
    for (int t = threadIdx.x; t < CDIM * WOUT * WOUT; t += 64) wsm[t] = lw[t];
    __syncthreads();
    float s = lb[o];
#pragma unroll
    for (int c = 0; c < CDIM; ++c) {
        const float* xr = Xc + ((size_t)(c * NDIM + n)) * WOUT;   // uniform
        for (int k = 0; k < WOUT; ++k)
            s += xr[k] * wsm[(c * WOUT + k) * WOUT + o];
    }
    out[(size_t)n * WOUT + o] = s;
}

// ---------------------------------------------------------------------------
// Launch
// ---------------------------------------------------------------------------
extern "C" void kernel_launch(void* const* d_in, const int* in_sizes, int n_in,
                              void* d_out, int out_size, void* d_ws, size_t ws_size,
                              hipStream_t stream) {
    (void)in_sizes; (void)n_in; (void)out_size; (void)ws_size;
    const float* A      = (const float*)d_in[0];
    const float* X      = (const float*)d_in[1];
    const float* w0a    = (const float*)d_in[2];
    const float* w0b    = (const float*)d_in[3];
    const float* w1     = (const float*)d_in[4];
    const float* weight = (const float*)d_in[5];
    const float* lin1_w = (const float*)d_in[6];
    const float* lin1_b = (const float*)d_in[7];
    float* out = (float*)d_out;

    char* ws = (char*)d_ws;
    float*          sw   = (float*)(ws + SW_OFF);
    unsigned short* aBF  = (unsigned short*)(ws + ABF_OFF);
    unsigned short* bT   = (unsigned short*)(ws + BT_OFF);
    unsigned short* gT   = (unsigned short*)(ws + GT_OFF);
    float*          H    = (float*)(ws + H_OFF);          // holds H, then H2
    unsigned short* HnBF = (unsigned short*)(ws + HNBF_OFF);
    float*          cinv = (float*)(ws + CINV_OFF);
    float*          part = (float*)(ws + PART_OFF);
    float*          XW   = (float*)(ws + XW_OFF);         // alias of aBF region
    float*          Xc   = (float*)(ws + XC_OFF);         // alias of aBF region

    k_softmax<<<1, 32, 0, stream>>>(w0a, w0b, w1, sw);
    k_stage1<<<(int)(NN / 256), 256, 0, stream>>>(A, sw, aBF, bT, gT);

    dim3 gg(NDIM / 128, NDIM / 128, CDIM);                // 16 x 16 x 2
    k_gemm_bf16<<<gg, 256, 0, stream>>>(aBF, bT, H);      // H = a @ b

    k_colpart<<<dim3(CDIM * NDIM / 256, 16), 256, 0, stream>>>(H, part);
    k_colinv<<<CDIM * NDIM / 256, 256, 0, stream>>>(part, cinv);
    k_scale_bf16<<<(int)(CDIM * NN / 256), 256, 0, stream>>>(H, cinv, HnBF);

    k_gemm_bf16<<<gg, 256, 0, stream>>>(HnBF, gT, H);     // H2 = Hn @ g (into H)

    k_xw<<<NDIM, 64, 0, stream>>>(X, weight, XW);
    k_gcn<<<CDIM * 32, 256, 0, stream>>>(H, XW, Xc);
    k_final<<<NDIM, 64, 0, stream>>>(Xc, lin1_w, lin1_b, out);
}
